// Mamba2Layer_84439057039491
// MI455X (gfx1250) — compile-verified, matrix operand order checked by reading, not run on hardware
//
#include <hip/hip_runtime.h>
#include <hip/hip_bf16.h>

typedef __attribute__((ext_vector_type(16))) __bf16 v16bf;
typedef __attribute__((ext_vector_type(8)))  __bf16 v8bf;
typedef __attribute__((ext_vector_type(8)))  float  v8f;

// ---------------- problem constants ----------------
constexpr int Lx    = 4096;
constexpr int Hx    = 2048;
constexpr int DIx   = 4096;
constexpr int Nx    = 128;
constexpr int NHx   = 64;
constexpr int Px    = 64;
constexpr int CHx   = 256;
constexpr int NCx   = 16;                 // L / CH
constexpr int CONVD = DIx + 2 * Nx;       // 4352
constexpr int NTOT  = CONVD + NHx;        // 4416

#define LDT 40   // LDS row stride in bf16 elements (32 data + 8 pad, keeps 16B align)

// ---------------- CDNA5 async global->LDS copy (ASYNCcnt-tracked) ----------------
#if defined(__has_builtin)
#if __has_builtin(__builtin_amdgcn_global_load_async_to_lds_b128)
#define HAVE_ASYNC 1
#endif
#endif

#ifdef HAVE_ASYNC
typedef int v4i __attribute__((__vector_size__(16)));
typedef __attribute__((address_space(1))) v4i as1_v4i;
typedef __attribute__((address_space(3))) v4i as3_v4i;
__device__ __forceinline__ void async_cp16(void* lds, const void* g) {
  __builtin_amdgcn_global_load_async_to_lds_b128((as1_v4i*)g, (as3_v4i*)lds, 0, 0);
}
#if __has_builtin(__builtin_amdgcn_s_wait_asynccnt)
#define WAIT_ASYNC(n) __builtin_amdgcn_s_wait_asynccnt(n)
#else
#define WAIT_ASYNC(n) asm volatile("s_wait_asynccnt %0" ::"i"(n))
#endif
#endif

// ---------------- scalar helpers ----------------
__device__ __forceinline__ unsigned short f2b(float f) {
  union { float f; unsigned u; } v; v.f = f;
  unsigned u = v.u;
  u += 0x7FFFu + ((u >> 16) & 1u);        // round-to-nearest-even
  return (unsigned short)(u >> 16);
}
__device__ __forceinline__ float b2f(unsigned short h) {
  union { unsigned u; float f; } v; v.u = ((unsigned)h) << 16; return v.f;
}

union U8 { unsigned short u[8]; v8bf v; };

// Load one 16x16x32 bf16 WMMA operand fragment from an LDS tile.
// Per-lane K mapping: e0..7 -> K = laneHi*8 + e, e8..15 -> K = 16 + laneHi*8 + (e-8).
__device__ __forceinline__ v16bf ldfrag(const unsigned short* base, int row, int hi8) {
  union { v16bf v; v8bf h[2]; } u;
  const unsigned short* p = base + row * LDT + hi8;
  u.h[0] = *(const v8bf*)(p);
  u.h[1] = *(const v8bf*)(p + 16);
  return u.v;
}

#define WMMA_BF16(A, B, C) \
  __builtin_amdgcn_wmma_f32_16x16x32_bf16(false, (A), false, (B), (short)0, (C), false, false)

// ---------------- fp32 -> bf16 convert ----------------
__global__ void cvt_f32_bf16(const float* __restrict__ src, unsigned short* __restrict__ dst,
                             long long n) {
  long long i = (long long)blockIdx.x * blockDim.x + threadIdx.x;
  long long stride = (long long)gridDim.x * blockDim.x;
  for (; i < n; i += stride) dst[i] = f2b(src[i]);
}

// ---------------- batched GEMM: C[m,n] = sum_k A[m,k] * B[n,k] ----------------
// A: (M,K) bf16 row-major, B: (N,K) bf16 row-major, C: (M,N) f32.
// Block tile 64x128, BK=32, 8 waves (2x4), each wave 32x32 = 2x2 WMMA frags.
// Async path: double-buffered LDS, GLOBAL_LOAD_ASYNC_TO_LDS_B128 prefetch of tile k+1
// overlapped with WMMA on tile k (ASYNCcnt in-order completion => wait<=3 releases buf k).
__global__ __launch_bounds__(256) void gemm_nt_bf16(
    const unsigned short* __restrict__ A, const unsigned short* __restrict__ B,
    float* __restrict__ C, int M, int N, int Kd,
    long long sA, long long sB, long long sC) {
  __shared__ __align__(16) unsigned short As[2][64 * LDT];
  __shared__ __align__(16) unsigned short Bs[2][128 * LDT];
  A += (size_t)blockIdx.z * sA;
  B += (size_t)blockIdx.z * sB;
  C += (size_t)blockIdx.z * sC;

  const int tid = threadIdx.x, lane = tid & 31, wave = tid >> 5;
  const int m0 = blockIdx.x * 64, n0 = blockIdx.y * 128;
  const int wm = (wave >> 2) * 32, wn = (wave & 3) * 32;
  const int lr = lane & 15, hi8 = (lane >> 4) << 3, hiAdd = (lane >> 4) << 3;

  const int arow = tid >> 2, akq = (tid & 3) << 3;   // A: 64 rows x 4 groups of 8
  const int brow = tid >> 1, bkq = (tid & 1) << 4;   // B: 128 rows x 2 groups of 16

  v8f acc[2][2] = {};
  const int nsteps = Kd >> 5;

#ifdef HAVE_ASYNC
  // M is always a multiple of 64 in our launches; clamp B edge rows (their columns
  // are discarded by the guarded store anyway).
  int gnB = n0 + brow;
  if (gnB >= N) gnB = N - 1;
  const unsigned short* gA = A + (size_t)(m0 + arow) * Kd + akq;
  const unsigned short* gB = B + (size_t)gnB * Kd + bkq;

  auto stage_async = [&](int buf, int k0) {
    async_cp16(&As[buf][arow * LDT + akq], gA + k0);
    async_cp16(&Bs[buf][brow * LDT + bkq], gB + k0);
    async_cp16(&Bs[buf][brow * LDT + bkq + 8], gB + k0 + 8);
  };
  stage_async(0, 0);
  for (int i = 0; i < nsteps; ++i) {
    if (i + 1 < nsteps) {
      stage_async((i + 1) & 1, (i + 1) << 5);
      WAIT_ASYNC(3);   // 3 newest (next buf) may fly; current buf complete
    } else {
      WAIT_ASYNC(0);
    }
    __syncthreads();   // all waves' copies for current buf landed
    const unsigned short* as = As[i & 1];
    const unsigned short* bs = Bs[i & 1];
    v16bf a0 = ldfrag(as, wm + lr, hi8);
    v16bf a1 = ldfrag(as, wm + 16 + lr, hi8);
    v16bf b0 = ldfrag(bs, wn + lr, hi8);
    v16bf b1 = ldfrag(bs, wn + 16 + lr, hi8);
    acc[0][0] = WMMA_BF16(a0, b0, acc[0][0]);
    acc[0][1] = WMMA_BF16(a0, b1, acc[0][1]);
    acc[1][0] = WMMA_BF16(a1, b0, acc[1][0]);
    acc[1][1] = WMMA_BF16(a1, b1, acc[1][1]);
    __syncthreads();   // done reading buf before it is re-staged at i+1
  }
#else
  for (int i = 0; i < nsteps; ++i) {
    int k0 = i << 5;
    { // stage A tile
      int gm = m0 + arow;
      v8bf v = {};
      if (gm < M) v = *(const v8bf*)(A + (size_t)gm * Kd + k0 + akq);
      *(v8bf*)(&As[0][arow * LDT + akq]) = v;
    }
    { // stage B tile
      int gn = n0 + brow;
      v8bf v0 = {}, v1 = {};
      if (gn < N) {
        const unsigned short* p = B + (size_t)gn * Kd + k0 + bkq;
        v0 = *(const v8bf*)(p);
        v1 = *(const v8bf*)(p + 8);
      }
      *(v8bf*)(&Bs[0][brow * LDT + bkq]) = v0;
      *(v8bf*)(&Bs[0][brow * LDT + bkq + 8]) = v1;
    }
    __syncthreads();
    v16bf a0 = ldfrag(As[0], wm + lr, hi8);
    v16bf a1 = ldfrag(As[0], wm + 16 + lr, hi8);
    v16bf b0 = ldfrag(Bs[0], wn + lr, hi8);
    v16bf b1 = ldfrag(Bs[0], wn + 16 + lr, hi8);
    acc[0][0] = WMMA_BF16(a0, b0, acc[0][0]);
    acc[0][1] = WMMA_BF16(a0, b1, acc[0][1]);
    acc[1][0] = WMMA_BF16(a1, b0, acc[1][0]);
    acc[1][1] = WMMA_BF16(a1, b1, acc[1][1]);
    __syncthreads();
  }
#endif

#pragma unroll
  for (int fi = 0; fi < 2; ++fi) {
    int mBase = m0 + wm + fi * 16;
#pragma unroll
    for (int fj = 0; fj < 2; ++fj) {
      int nCol = n0 + wn + fj * 16 + lr;
      if (nCol < N) {
        v8f a = acc[fi][fj];
#pragma unroll
        for (int r = 0; r < 8; ++r) {
          int m = mBase + r + hiAdd;
          if (m < M) C[(size_t)m * N + nCol] = a[r];
        }
      }
    }
  }
}

// ---------------- depthwise causal conv(K=4) + SiLU ----------------
__global__ void conv_silu_k(const float* __restrict__ xBCdt,
                            const float* __restrict__ w, const float* __restrict__ b,
                            float* __restrict__ Xssm,
                            unsigned short* __restrict__ Bb,
                            unsigned short* __restrict__ Btb,
                            unsigned short* __restrict__ Cb) {
  long long idx = (long long)blockIdx.x * blockDim.x + threadIdx.x;
  if (idx >= (long long)Lx * CONVD) return;
  int t = (int)(idx / CONVD);
  int ch = (int)(idx % CONVD);
  float acc = b[ch];
#pragma unroll
  for (int j = 0; j < 4; ++j) {
    int tt = t - 3 + j;
    if (tt >= 0) acc += w[ch * 4 + j] * xBCdt[(size_t)tt * NTOT + ch];
  }
  float s = acc / (1.f + __expf(-acc));       // SiLU
  if (ch < DIx) {
    Xssm[(size_t)t * DIx + ch] = s;
  } else if (ch < DIx + Nx) {
    int n = ch - DIx;
    unsigned short hb = f2b(s);
    Bb[(size_t)t * Nx + n] = hb;
    Btb[(size_t)n * Lx + t] = hb;
  } else {
    int n = ch - DIx - Nx;
    Cb[(size_t)t * Nx + n] = f2b(s);
  }
}

// ---------------- dt = clip(softplus(raw + bias)), dA = dt * (-exp(A_log)) ----------------
__global__ void dt_k(const float* __restrict__ xBCdt, const float* __restrict__ dt_bias,
                     const float* __restrict__ A_log,
                     float* __restrict__ dtv, float* __restrict__ dAv) {
  int idx = blockIdx.x * blockDim.x + threadIdx.x;   // L*NH
  if (idx >= Lx * NHx) return;
  int t = idx >> 6, h = idx & 63;
  float x = xBCdt[(size_t)t * NTOT + CONVD + h] + dt_bias[h];
  float sp = (x > 20.f) ? x : log1pf(__expf(x));
  sp = fminf(fmaxf(sp, 0.001f), 100.f);
  dtv[idx] = sp;
  dAv[idx] = -sp * __expf(A_log[h]);
}

// ---------------- Xdt transposed bf16: (NH*P, L), Xdt_t[i*L+t] = Xssm[t,i]*dt[t,h] ----------------
__global__ void scalex_k(const float* __restrict__ Xssm, const float* __restrict__ dtv,
                         unsigned short* __restrict__ Xdt_t) {
  long long idx = (long long)blockIdx.x * blockDim.x + threadIdx.x;  // DI*L
  if (idx >= (long long)DIx * Lx) return;
  int i = (int)(idx >> 12);    // feature (h*64+p)
  int t = (int)(idx & (Lx - 1));
  int h = i >> 6;
  Xdt_t[idx] = f2b(Xssm[(size_t)t * DIx + i] * dtv[t * NHx + h]);
}

// ---------------- per-(chunk,head) inclusive cumsum of dA ----------------
__global__ void cumsum_k(const float* __restrict__ dAv, float* __restrict__ Acs,
                         float* __restrict__ cs) {
  int c = blockIdx.x >> 6, h = blockIdx.x & 63;
  int i = threadIdx.x;   // 0..255
  __shared__ float s[CHx];
  s[i] = dAv[(size_t)(c * CHx + i) * NHx + h];
  __syncthreads();
  for (int off = 1; off < CHx; off <<= 1) {
    float add = (i >= off) ? s[i - off] : 0.f;
    __syncthreads();
    s[i] += add;
    __syncthreads();
  }
  Acs[(size_t)(c * CHx + i) * NHx + h] = s[i];
  if (i == 255) cs[c * NHx + h] = s[i];
}

// ---------------- states[c,h][n,p] = sum_k B[c,k,n]*exp(cs_last - Acs_k)*Xdt[c,k,h,p] ----------------
// WMMA: A[n,k] = Bt*decay (bf16), B[p,k] = Xdt_t.  Out 128x64, K=256. grid (nc*NH, 2).
__global__ __launch_bounds__(256) void states_k(
    const unsigned short* __restrict__ Btb, const unsigned short* __restrict__ Xdt_t,
    const float* __restrict__ Acs, float* __restrict__ states) {
  int ch = blockIdx.x;
  int c = ch >> 6, h = ch & 63;
  int mtile = blockIdx.y;   // 0..1 -> rows of n
  __shared__ __align__(16) unsigned short As[64 * LDT];
  __shared__ __align__(16) unsigned short Bs[64 * LDT];
  __shared__ float dcol[32];

  const int tid = threadIdx.x, lane = tid & 31, wave = tid >> 5;
  const int wm = (wave >> 2) * 32, wn = (wave & 3) * 16;
  const int lr = lane & 15, hi8 = (lane >> 4) << 3, hiAdd = (lane >> 4) << 3;
  const int srow = tid >> 2, skq = (tid & 3) << 3;

  const float csLast = Acs[(size_t)(c * CHx + CHx - 1) * NHx + h];
  v8f acc[2] = {};

  for (int k0 = 0; k0 < CHx; k0 += 32) {
    if (tid < 32)
      dcol[tid] = __expf(csLast - Acs[(size_t)(c * CHx + k0 + tid) * NHx + h]);
    __syncthreads();
    { // A: rows n, cols k, scaled by decay
      int ng = mtile * 64 + srow;
      const unsigned short* bp = Btb + (size_t)ng * Lx + c * CHx + k0 + skq;
      U8 tu;
#pragma unroll
      for (int j = 0; j < 8; ++j) tu.u[j] = f2b(b2f(bp[j]) * dcol[skq + j]);
      *(v8bf*)(As + srow * LDT + skq) = tu.v;
    }
    { // B: rows p, cols k (already bf16, contiguous) -> async copy straight to LDS
      const unsigned short* gp = Xdt_t + (size_t)(h * Px + srow) * Lx + c * CHx + k0 + skq;
#ifdef HAVE_ASYNC
      async_cp16(Bs + srow * LDT + skq, gp);
      WAIT_ASYNC(0);
#else
      *(v8bf*)(Bs + srow * LDT + skq) = *(const v8bf*)gp;
#endif
    }
    __syncthreads();
    v16bf a0 = ldfrag(As, wm + lr, hi8);
    v16bf a1 = ldfrag(As, wm + 16 + lr, hi8);
    v16bf b0 = ldfrag(Bs, wn + lr, hi8);
    acc[0] = WMMA_BF16(a0, b0, acc[0]);
    acc[1] = WMMA_BF16(a1, b0, acc[1]);
  }

  int p = wn + lr;
#pragma unroll
  for (int fi = 0; fi < 2; ++fi) {
    v8f a = acc[fi];
#pragma unroll
    for (int r = 0; r < 8; ++r) {
      int n = mtile * 64 + wm + fi * 16 + r + hiAdd;
      states[(size_t)ch * (Nx * Px) + (size_t)n * Px + p] = a[r];
    }
  }
}

// ---------------- inter-chunk recurrence: prev[z] = exp(cs[z-1])*prev[z-1] + states[z-1] ----------------
// states layout (c,h,n,p); prev written transposed as (c,h,p,n).
__global__ void prev_k(const float* __restrict__ states, const float* __restrict__ cs,
                       float* __restrict__ prev) {
  int h = blockIdx.x;
  for (int e = threadIdx.x; e < Nx * Px; e += blockDim.x) {
    int n = e >> 6, p = e & 63;
    float r = 0.f;
    for (int z = 0; z < NCx; ++z) {
      prev[(size_t)(z * NHx + h) * (Nx * Px) + (size_t)p * Nx + n] = r;
      r = __expf(cs[z * NHx + h]) * r + states[(size_t)(z * NHx + h) * (Nx * Px) + e];
    }
  }
}

// ---------------- Y[c,k,h,p] = sum_s G[k,s]e^{Acs_k-Acs_s}Xdt[s,h,p]
//                             + sum_n C[k,n]e^{Acs_k}prev[h,p,n]  + D[h]*Xssm ----------------
// Out tile 64(k) x 64(p) per block; grid (nc*NH, 4).
__global__ __launch_bounds__(256) void ydiag_k(
    const float* __restrict__ G, const float* __restrict__ Acs,
    const unsigned short* __restrict__ Xdt_t, const unsigned short* __restrict__ Cb,
    const float* __restrict__ prev, const float* __restrict__ Xssm,
    const float* __restrict__ Dp, float* __restrict__ Y) {
  int chb = blockIdx.x;
  int c = chb >> 6, h = chb & 63;
  int mtile = blockIdx.y;  // 0..3, 64 rows of k each
  __shared__ __align__(16) unsigned short As[64 * LDT];
  __shared__ __align__(16) unsigned short Bs[64 * LDT];
  __shared__ float acsRow[64];
  __shared__ float acsCol[32];

  const int tid = threadIdx.x, lane = tid & 31, wave = tid >> 5;
  const int wm = (wave >> 2) * 32, wn = (wave & 3) * 16;
  const int lr = lane & 15, hi8 = (lane >> 4) << 3, hiAdd = (lane >> 4) << 3;
  const int srow = tid >> 2, skq = (tid & 3) << 3;

  if (tid < 64)
    acsRow[tid] = Acs[(size_t)(c * CHx + mtile * 64 + tid) * NHx + h];

  v8f acc[2] = {};

  // ---- phase 1: (G * decay-mask) @ Xdt, K over s = 0..255 ----
  for (int k0 = 0; k0 < CHx; k0 += 32) {
    if (tid < 32)
      acsCol[tid] = Acs[(size_t)(c * CHx + k0 + tid) * NHx + h];
    __syncthreads();   // acsCol/acsRow visible; prior frag reads done
    { // A: S[k,s] = G[k,s]*exp(Acs_k - Acs_s) for s<=k else 0
      int kg = mtile * 64 + srow;
      float ak = acsRow[srow];
      const float* gp = G + ((size_t)c << 16) + (size_t)kg * CHx + k0 + skq;
      U8 tu;
#pragma unroll
      for (int j = 0; j < 8; ++j) {
        int s = k0 + skq + j;
        float v = (s <= kg) ? gp[j] * __expf(ak - acsCol[skq + j]) : 0.f;
        tu.u[j] = f2b(v);
      }
      *(v8bf*)(As + srow * LDT + skq) = tu.v;
    }
    { // B: rows p, cols s -> async copy straight to LDS
      const unsigned short* gp = Xdt_t + (size_t)(h * Px + srow) * Lx + c * CHx + k0 + skq;
#ifdef HAVE_ASYNC
      async_cp16(Bs + srow * LDT + skq, gp);
      WAIT_ASYNC(0);
#else
      *(v8bf*)(Bs + srow * LDT + skq) = *(const v8bf*)gp;
#endif
    }
    __syncthreads();
    v16bf a0 = ldfrag(As, wm + lr, hi8);
    v16bf a1 = ldfrag(As, wm + 16 + lr, hi8);
    v16bf b0 = ldfrag(Bs, wn + lr, hi8);
    acc[0] = WMMA_BF16(a0, b0, acc[0]);
    acc[1] = WMMA_BF16(a1, b0, acc[1]);
  }

  // ---- phase 2: (C * e^{Acs_k}) @ prev^T, K over n = 0..127 ----
  for (int k0 = 0; k0 < Nx; k0 += 32) {
    __syncthreads();   // protect As/Bs from prior frag reads
    {
      int kg = mtile * 64 + srow;
      float ek = __expf(acsRow[srow]);
      const unsigned short* cp = Cb + (size_t)(c * CHx + kg) * Nx + k0 + skq;
      U8 tu;
#pragma unroll
      for (int j = 0; j < 8; ++j) tu.u[j] = f2b(b2f(cp[j]) * ek);
      *(v8bf*)(As + srow * LDT + skq) = tu.v;
    }
    {
      const float* pp = prev + (size_t)chb * (Nx * Px) + (size_t)srow * Nx + k0 + skq;
      U8 tu;
#pragma unroll
      for (int j = 0; j < 8; ++j) tu.u[j] = f2b(pp[j]);
      *(v8bf*)(Bs + srow * LDT + skq) = tu.v;
    }
    __syncthreads();
    v16bf a0 = ldfrag(As, wm + lr, hi8);
    v16bf a1 = ldfrag(As, wm + 16 + lr, hi8);
    v16bf b0 = ldfrag(Bs, wn + lr, hi8);
    acc[0] = WMMA_BF16(a0, b0, acc[0]);
    acc[1] = WMMA_BF16(a1, b0, acc[1]);
  }

  // ---- epilogue: + D[h]*Xssm, write Y ----
  float dh = Dp[h];
  int p = wn + lr;
#pragma unroll
  for (int fi = 0; fi < 2; ++fi) {
    v8f a = acc[fi];
#pragma unroll
    for (int r = 0; r < 8; ++r) {
      int kloc = mtile * 64 + wm + fi * 16 + r + hiAdd;
      size_t o = (size_t)(c * CHx + kloc) * DIx + h * Px + p;
      Y[o] = a[r] + dh * Xssm[o];
    }
  }
}

// ---------------- RMSNorm over DI, write bf16 ----------------
__global__ void norm_k(const float* __restrict__ Y, const float* __restrict__ nw,
                       unsigned short* __restrict__ yn) {
  int t = blockIdx.x, tid = threadIdx.x;
  __shared__ float red[256];
  float ss = 0.f;
  for (int i = tid; i < DIx; i += 256) {
    float v = Y[(size_t)t * DIx + i];
    ss += v * v;
  }
  red[tid] = ss;
  __syncthreads();
  for (int off = 128; off > 0; off >>= 1) {
    if (tid < off) red[tid] += red[tid + off];
    __syncthreads();
  }
  float rs = rsqrtf(red[0] / (float)DIx + 1e-6f);
  for (int i = tid; i < DIx; i += 256)
    yn[(size_t)t * DIx + i] = f2b(Y[(size_t)t * DIx + i] * rs * nw[i]);
}

// ---------------- host launcher ----------------
extern "C" void kernel_launch(void* const* d_in, const int* in_sizes, int n_in,
                              void* d_out, int out_size, void* d_ws, size_t ws_size,
                              hipStream_t stream) {
  const float* x       = (const float*)d_in[0];
  const float* W_in    = (const float*)d_in[1];
  const float* conv_w  = (const float*)d_in[2];
  const float* conv_b  = (const float*)d_in[3];
  const float* dt_bias = (const float*)d_in[4];
  const float* A_log   = (const float*)d_in[5];
  const float* Dvec    = (const float*)d_in[6];
  const float* norm_w  = (const float*)d_in[7];
  const float* W_out   = (const float*)d_in[8];
  float* out = (float*)d_out;

  char* w = (char*)d_ws;
  size_t off = 0;
  auto alloc = [&](size_t bytes) -> void* {
    void* p = w + off;
    off = (off + bytes + 255) & ~(size_t)255;
    return p;
  };

  unsigned short* xh    = (unsigned short*)alloc((size_t)Lx * Hx * 2);
  unsigned short* WinH  = (unsigned short*)alloc((size_t)NTOT * Hx * 2);
  unsigned short* WoutH = (unsigned short*)alloc((size_t)Hx * DIx * 2);
  float* xBCdt = (float*)alloc((size_t)Lx * NTOT * 4);   // later reused as Y
  float* Xssm  = (float*)alloc((size_t)Lx * DIx * 4);
  unsigned short* Bb  = (unsigned short*)alloc((size_t)Lx * Nx * 2);
  unsigned short* Btb = (unsigned short*)alloc((size_t)Nx * Lx * 2);
  unsigned short* Cb  = (unsigned short*)alloc((size_t)Lx * Nx * 2);
  float* dtv = (float*)alloc((size_t)Lx * NHx * 4);
  float* dAv = (float*)alloc((size_t)Lx * NHx * 4);
  float* Acs = (float*)alloc((size_t)Lx * NHx * 4);
  float* cs  = (float*)alloc((size_t)NCx * NHx * 4);
  float* G   = (float*)alloc((size_t)NCx * CHx * CHx * 4);
  unsigned short* Xdt_t = (unsigned short*)alloc((size_t)DIx * Lx * 2);
  float* states = (float*)alloc((size_t)NCx * NHx * Nx * Px * 4);
  float* prev   = (float*)alloc((size_t)NCx * NHx * Nx * Px * 4);
  unsigned short* yn = (unsigned short*)alloc((size_t)Lx * DIx * 2);
  float* Y = xBCdt;   // reuse: xBCdt fully consumed before Y is produced

  // 1. bf16 conversions
  cvt_f32_bf16<<<2048, 256, 0, stream>>>(x, xh, (long long)Lx * Hx);
  cvt_f32_bf16<<<2048, 256, 0, stream>>>(W_in, WinH, (long long)NTOT * Hx);
  cvt_f32_bf16<<<2048, 256, 0, stream>>>(W_out, WoutH, (long long)Hx * DIx);

  // 2. input GEMM: xBCdt = x @ W_in^T   (M=4096, N=4416, K=2048)
  gemm_nt_bf16<<<dim3(Lx / 64, (NTOT + 127) / 128, 1), 256, 0, stream>>>(
      xh, WinH, xBCdt, Lx, NTOT, Hx, 0, 0, 0);

  // 3. conv + SiLU; dt
  conv_silu_k<<<(int)(((long long)Lx * CONVD + 255) / 256), 256, 0, stream>>>(
      xBCdt, conv_w, conv_b, Xssm, Bb, Btb, Cb);
  dt_k<<<(Lx * NHx + 255) / 256, 256, 0, stream>>>(xBCdt, dt_bias, A_log, dtv, dAv);

  // 4. Xdt transpose-scale, cumsum
  scalex_k<<<(int)(((long long)DIx * Lx + 255) / 256), 256, 0, stream>>>(Xssm, dtv, Xdt_t);
  cumsum_k<<<NCx * NHx, CHx, 0, stream>>>(dAv, Acs, cs);

  // 5. per-chunk G = C @ B^T  (M=N=256, K=128, batch=16)
  gemm_nt_bf16<<<dim3(CHx / 64, CHx / 128, NCx), 256, 0, stream>>>(
      Cb, Bb, G, CHx, CHx, Nx,
      (long long)CHx * Nx, (long long)CHx * Nx, (long long)CHx * CHx);

  // 6. chunk states + inter-chunk recurrence
  states_k<<<dim3(NCx * NHx, 2), 256, 0, stream>>>(Btb, Xdt_t, Acs, states);
  prev_k<<<NHx, 256, 0, stream>>>(states, cs, prev);

  // 7. Y = Y_diag + Y_off + D*x_ssm
  ydiag_k<<<dim3(NCx * NHx, 4), 256, 0, stream>>>(G, Acs, Xdt_t, Cb, prev, Xssm, Dvec, Y);

  // 8. RMSNorm -> bf16
  norm_k<<<Lx, 256, 0, stream>>>(Y, norm_w, yn);

  // 9. output GEMM: out = y_norm @ W_out^T  (M=4096, N=2048, K=4096)
  gemm_nt_bf16<<<dim3(Lx / 64, Hx / 128, 1), 256, 0, stream>>>(
      yn, WoutH, out, Lx, Hx, DIx, 0, 0, 0);
}
